// SparseMoeBlock_50646254355306
// MI455X (gfx1250) — compile-verified, matrix operand order checked by reading, not run on hardware
//
#include <hip/hip_runtime.h>
#include <math.h>

// ---------------- problem constants (match reference) ----------------
#define T_TOK 2048   // B*S tokens
#define DDIM  2048   // hidden dim
#define FDIM  512    // moe intermediate
#define NEXP  16     // experts
#define TOPK  4      // experts per token

typedef __attribute__((ext_vector_type(16))) __bf16 v16bf;
typedef __attribute__((ext_vector_type(8)))  __bf16 v8bf;
typedef __attribute__((ext_vector_type(8)))  float  v8f;

// round-to-nearest-even fp32 -> bf16
__device__ __forceinline__ unsigned short f2bf_bits(float x) {
  unsigned u = __float_as_uint(x);
  return (unsigned short)((u + 0x7FFFu + ((u >> 16) & 1u)) >> 16);
}
__device__ __forceinline__ __bf16 f2bf(float x) {
  unsigned short b = f2bf_bits(x);
  __bf16 h;
  __builtin_memcpy(&h, &b, 2);
  return h;
}
__device__ __forceinline__ v8f vzero8() {
  v8f z;
#pragma unroll
  for (int i = 0; i < 8; ++i) z[i] = 0.f;
  return z;
}
__device__ __forceinline__ v8f wmma_bf16(v16bf a, v16bf b, v8f c) {
  // D = A(16x32 bf16) x B(32x16 bf16) + C(16x16 f32)
  return __builtin_amdgcn_wmma_f32_16x16x32_bf16(false, a, false, b,
                                                 (short)0, c, false, false);
}
__device__ __forceinline__ float fast_sigmoid(float x) {
  return __builtin_amdgcn_rcpf(1.f + __expf(-x));  // v_rcp_f32, no div expansion
}

// ---------------- kernel 0: zero per-expert counters ----------------
__global__ void moe_zero_cnt(int* __restrict__ cnt) {
  if (threadIdx.x < NEXP) cnt[threadIdx.x] = 0;
}

// ---------------- kernel R: repack fp32 weights -> bf16 WMMA-frag layout ----
// src: nmat row-major [K][N] fp32 matrices. dst layout per matrix:
//   [kc = k/32][ft = n/16][lane = (n&15) + 16*hi][j = (k&7) + 8*(k>=16 in chunk)]
// so a lane's 16 K-values for one (kc,ft) B-tile are 32 contiguous bytes.
// Each thread converts 8 consecutive k for one n -> one 16B packed store.
__global__ void repack_bf16(const float* __restrict__ src,
                            unsigned short* __restrict__ dst,
                            int nShift, int kShift) {
  int i = blockIdx.x * 256 + threadIdx.x;   // i = ((mat*(K/8)+k8)*N + n)
  const int N = 1 << nShift;
  int n  = i & (N - 1);
  int t2 = i >> nShift;
  int k8 = t2 & ((1 << (kShift - 3)) - 1);
  int mat = t2 >> (kShift - 3);
  int k0 = k8 << 3;
  size_t ms = (size_t)mat << (kShift + nShift);
  const float* s = src + ms + (size_t)k0 * N + n;
  int dk = k0 & 31, kc = k0 >> 5;
  int lane = (n & 15) + (((dk >> 3) & 1) << 4);
  int ft = n >> 4;
  int jb = (dk & 16) >> 1;
  unsigned short* d =
      dst + ms + ((((size_t)kc * (N >> 4) + ft) * 32 + lane) << 4) + jb;
  v8bf v;
#pragma unroll
  for (int j = 0; j < 8; ++j) v[j] = f2bf(s[(size_t)j * N]);
  *(v8bf*)d = v;
}

// ---------------- kernel 1: router (one wave32 per token) ----------------
__global__ void moe_router(const float* __restrict__ x,    // [T,D]
                           const float* __restrict__ gw,   // [E,D]
                           const float* __restrict__ sgw,  // [D,1]
                           int* __restrict__ cnt,          // [E]
                           int* __restrict__ tokL,         // [E,T]
                           float* __restrict__ wL,         // [E,T]
                           float* __restrict__ sgate)      // [T]
{
  const int lane = threadIdx.x & 31;
  const int wv   = threadIdx.x >> 5;
  const int t    = blockIdx.x * 8 + wv;

  float acc[NEXP];
#pragma unroll
  for (int e = 0; e < NEXP; ++e) acc[e] = 0.f;
  float sg = 0.f;

  const float* xr = x + (size_t)t * DDIM;
  for (int d = lane; d < DDIM; d += 32) {
    float xv = xr[d];
#pragma unroll
    for (int e = 0; e < NEXP; ++e) acc[e] += xv * gw[e * DDIM + d];
    sg += xv * sgw[d];
  }
#pragma unroll
  for (int off = 16; off > 0; off >>= 1) {
#pragma unroll
    for (int e = 0; e < NEXP; ++e) acc[e] += __shfl_xor(acc[e], off, 32);
    sg += __shfl_xor(sg, off, 32);
  }

  // top-4 on logits == top-4 on softmax probs; renormalize among the 4
  float q[NEXP];
#pragma unroll
  for (int e = 0; e < NEXP; ++e) q[e] = acc[e];
  int esel[TOPK];
  float lsel[TOPK];
#pragma unroll
  for (int k = 0; k < TOPK; ++k) {
    float best = -1e30f;
    int bj = 0;
#pragma unroll
    for (int e = 0; e < NEXP; ++e)
      if (q[e] > best) { best = q[e]; bj = e; }
    esel[k] = bj;
    lsel[k] = best;
#pragma unroll
    for (int e = 0; e < NEXP; ++e)
      if (e == bj) q[e] = -1e30f;
  }
  float wk[TOPK], wsum = 0.f;
#pragma unroll
  for (int k = 0; k < TOPK; ++k) { wk[k] = __expf(lsel[k] - lsel[0]); wsum += wk[k]; }

  if (lane == 0) {
    float inv = __builtin_amdgcn_rcpf(wsum);
#pragma unroll
    for (int k = 0; k < TOPK; ++k) {
      int e = esel[k];
      int pos = atomicAdd(&cnt[e], 1);
      tokL[e * T_TOK + pos] = t;
      wL[e * T_TOK + pos] = wk[k] * inv;
    }
    sgate[t] = fast_sigmoid(sg);
  }
}

// ---------------- kernel 2/3: swiglu FFN over a 16-token tile -------------
// 256 threads = 8 waves. Wave wv owns F-tiles [wv*4, wv*4+4) for gate/up and
// D-tiles [wv*16, wv*16+16) (2 passes of 8) for down-proj.
// LDS holds activations/intermediate already in WMMA A-fragment layout:
//   s[kc][lane][j], lane = row + 16*hi(K), j = (k&7)+8*(k chunk-high)
template <bool SHARED, bool PACKED>
__global__ __launch_bounds__(256, 1) void moe_ffn(
    const float* __restrict__ x,                                 // [T,D]
    const float* __restrict__ Wgp, const float* __restrict__ Wup,
    const float* __restrict__ Wdp,                               // fp32 fallback
    const unsigned short* __restrict__ pWg,
    const unsigned short* __restrict__ pWu,
    const unsigned short* __restrict__ pWd,                      // packed bf16
    const int* __restrict__ cnt, const int* __restrict__ tokL,
    const float* __restrict__ wL, const float* __restrict__ sgate,
    float* __restrict__ out)                                     // [T,D]
{
  const int e = SHARED ? 0 : blockIdx.y;
  const int tile = blockIdx.x;
  int ntok;
  if constexpr (SHARED) ntok = T_TOK; else ntok = cnt[e];
  if (tile * 16 >= ntok) return;  // uniform exit

  __shared__ __align__(16) unsigned short sA[DDIM / 32][32][16];  // 64KB
  __shared__ __align__(16) unsigned short sH[FDIM / 32][32][16];  // 16KB
  __shared__ int sT[16];
  __shared__ float sW[16];

  const int tid = threadIdx.x;
  if (tid < 16) {
    int idx = tile * 16 + tid;
    if constexpr (SHARED) {
      sT[tid] = idx;
      sW[tid] = sgate[idx];
    } else {
      if (idx < ntok) { sT[tid] = tokL[e * T_TOK + idx]; sW[tid] = wL[e * T_TOK + idx]; }
      else            { sT[tid] = 0;                     sW[tid] = 0.f; }  // padded row
    }
  }
  __syncthreads();

  // stage 16 gathered token rows as bf16 in frag layout (8 d per thread)
  for (int i = tid; i < 16 * (DDIM / 8); i += 256) {
    int m  = i >> 8;               // DDIM/8 = 256 groups per token
    int d0 = (i & 255) << 3;
    const float* p = x + (size_t)sT[m] * DDIM + d0;
    int dk = d0 & 31, kc = d0 >> 5;
    int lane_ = m + (((dk >> 3) & 1) << 4);
    int jb = (dk & 16) >> 1;
    v8bf v;
#pragma unroll
    for (int j = 0; j < 8; ++j) v[j] = f2bf(p[j]);
    *(v8bf*)&sA[kc][lane_][jb] = v;
  }
  __syncthreads();

  const int lane = tid & 31;
  const int wv   = tid >> 5;
  const int half = lane >> 4;
  const int col  = lane & 15;
  const int koff = half * 8;

  const float* Wge = Wgp ? Wgp + (size_t)e * DDIM * FDIM : nullptr;
  const float* Wue = Wup ? Wup + (size_t)e * DDIM * FDIM : nullptr;
  const float* Wde = Wdp ? Wdp + (size_t)e * FDIM * DDIM : nullptr;
  const unsigned short* pWge = pWg ? pWg + (size_t)e * DDIM * FDIM : nullptr;
  const unsigned short* pWue = pWu ? pWu + (size_t)e * DDIM * FDIM : nullptr;
  const unsigned short* pWde = pWd ? pWd + (size_t)e * FDIM * DDIM : nullptr;

  // ---- gate & up projections: C[16 x 64] per wave, K = D ----
  v8f accG[4], accU[4];
#pragma unroll
  for (int i = 0; i < 4; ++i) { accG[i] = vzero8(); accU[i] = vzero8(); }

  for (int kc = 0; kc < DDIM / 32; ++kc) {
    v16bf a = *(const v16bf*)&sA[kc][lane][0];
#pragma unroll
    for (int nt = 0; nt < 4; ++nt) {
      int ft = wv * 4 + nt;
      v16bf bg, bu;
      if constexpr (PACKED) {
        size_t off = (((size_t)kc * (FDIM >> 4) + ft) * 32 + lane) << 4;
        bg = *(const v16bf*)(pWge + off);
        bu = *(const v16bf*)(pWue + off);
      } else {
        int f = ft * 16 + col;
        int kb = kc * 32 + koff;
#pragma unroll
        for (int j = 0; j < 8; ++j) {
          bg[j]     = f2bf(Wge[(size_t)(kb + j) * FDIM + f]);
          bg[8 + j] = f2bf(Wge[(size_t)(kb + 16 + j) * FDIM + f]);
          bu[j]     = f2bf(Wue[(size_t)(kb + j) * FDIM + f]);
          bu[8 + j] = f2bf(Wue[(size_t)(kb + 16 + j) * FDIM + f]);
        }
      }
      accG[nt] = wmma_bf16(a, bg, accG[nt]);
      accU[nt] = wmma_bf16(a, bu, accU[nt]);
    }
  }

  // h = silu(g) * u * w_token (weight folded so down-proj output is pre-scaled)
#pragma unroll
  for (int nt = 0; nt < 4; ++nt) {
    int f = wv * 64 + nt * 16 + col;
    int kcf = f >> 5, dkf = f & 31;
    int laneH = (((dkf >> 3) & 1) << 4);
    int jf = (dkf & 7) + ((dkf & 16) >> 1);
#pragma unroll
    for (int r = 0; r < 8; ++r) {
      int row = r + half * 8;
      float g = accG[nt][r], u = accU[nt][r];
      float h = g * fast_sigmoid(g) * u * sW[row];
      sH[kcf][row + laneH][jf] = f2bf_bits(h);
    }
  }
  __syncthreads();

  // ---- down projection: out[16 x 2048], K = F ----
#pragma unroll 1
  for (int pass = 0; pass < 2; ++pass) {
    const int ftBase = wv * 16 + pass * 8;   // D-tiles of 16 cols
    v8f acc[8];
#pragma unroll
    for (int i = 0; i < 8; ++i) acc[i] = vzero8();

    for (int kc = 0; kc < FDIM / 32; ++kc) {
      v16bf a = *(const v16bf*)&sH[kc][lane][0];
#pragma unroll
      for (int nt = 0; nt < 8; ++nt) {
        int ft = ftBase + nt;
        v16bf b;
        if constexpr (PACKED) {
          size_t off = (((size_t)kc * (DDIM >> 4) + ft) * 32 + lane) << 4;
          b = *(const v16bf*)(pWde + off);
        } else {
          int d = ft * 16 + col;
          int kb = kc * 32 + koff;
#pragma unroll
          for (int j = 0; j < 8; ++j) {
            b[j]     = f2bf(Wde[(size_t)(kb + j) * DDIM + d]);
            b[8 + j] = f2bf(Wde[(size_t)(kb + 16 + j) * DDIM + d]);
          }
        }
        acc[nt] = wmma_bf16(a, b, acc[nt]);
      }
    }
#pragma unroll
    for (int nt = 0; nt < 8; ++nt) {
      int d = (ftBase + nt) * 16 + col;
#pragma unroll
      for (int r = 0; r < 8; ++r) {
        int row = r + half * 8;
        float v = acc[nt][r];
        float* dst = out + (size_t)sT[row] * DDIM + d;
        if constexpr (SHARED) *dst = v;          // initializes d_out
        else atomicAdd(dst, v);                  // experts accumulate on top
      }
    }
  }
}

// ---------------- host-side launch ----------------
extern "C" void kernel_launch(void* const* d_in, const int* in_sizes, int n_in,
                              void* d_out, int out_size, void* d_ws, size_t ws_size,
                              hipStream_t stream) {
  const float* hs  = (const float*)d_in[0];  // [1,2048,2048]
  const float* gw  = (const float*)d_in[1];  // [16,2048]
  const float* Wg  = (const float*)d_in[2];  // [16,2048,512]
  const float* Wu  = (const float*)d_in[3];  // [16,2048,512]
  const float* Wd  = (const float*)d_in[4];  // [16,512,2048]
  const float* sWg = (const float*)d_in[5];  // [2048,512]
  const float* sWu = (const float*)d_in[6];  // [2048,512]
  const float* sWd = (const float*)d_in[7];  // [512,2048]
  const float* sgw = (const float*)d_in[8];  // [2048,1]
  float* out = (float*)d_out;

  // workspace layout
  const size_t EW = (size_t)DDIM * FDIM;  // elems per expert matrix (1M)
  size_t off = 0;
  auto take = [&](size_t bytes) { size_t o = off; off += (bytes + 255) & ~(size_t)255; return o; };
  char* w = (char*)d_ws;
  int*   cnt   = (int*)(w + take(NEXP * sizeof(int)));
  int*   tokL  = (int*)(w + take((size_t)NEXP * T_TOK * sizeof(int)));
  float* wL    = (float*)(w + take((size_t)NEXP * T_TOK * sizeof(float)));
  float* sgate = (float*)(w + take(T_TOK * sizeof(float)));
  size_t smallNeed = off;
  unsigned short* pWg  = (unsigned short*)(w + take(NEXP * EW * 2));
  unsigned short* pWu  = (unsigned short*)(w + take(NEXP * EW * 2));
  unsigned short* pWd  = (unsigned short*)(w + take(NEXP * EW * 2));
  unsigned short* psWg = (unsigned short*)(w + take(EW * 2));
  unsigned short* psWu = (unsigned short*)(w + take(EW * 2));
  unsigned short* psWd = (unsigned short*)(w + take(EW * 2));
  const bool packed = (ws_size >= off) ;
  (void)smallNeed;

  moe_zero_cnt<<<1, 32, 0, stream>>>(cnt);
  moe_router<<<T_TOK / 8, 256, 0, stream>>>(hs, gw, sgw, cnt, tokL, wL, sgate);

  if (packed) {
    // repack all weights into bf16 WMMA-fragment layout (reads fp32 once)
    const int BG = (int)(NEXP * EW / 8 / 256);  // 8192 blocks for expert mats
    const int BS = (int)(EW / 8 / 256);         // 512 blocks for shared mats
    repack_bf16<<<BG, 256, 0, stream>>>(Wg,  pWg,  9, 11);   // K=2048,N=512
    repack_bf16<<<BG, 256, 0, stream>>>(Wu,  pWu,  9, 11);
    repack_bf16<<<BG, 256, 0, stream>>>(Wd,  pWd, 11,  9);   // K=512,N=2048
    repack_bf16<<<BS, 256, 0, stream>>>(sWg, psWg, 9, 11);
    repack_bf16<<<BS, 256, 0, stream>>>(sWu, psWu, 9, 11);
    repack_bf16<<<BS, 256, 0, stream>>>(sWd, psWd, 11, 9);

    moe_ffn<true, true><<<dim3(T_TOK / 16), 256, 0, stream>>>(
        hs, nullptr, nullptr, nullptr, psWg, psWu, psWd,
        nullptr, nullptr, nullptr, sgate, out);
    moe_ffn<false, true><<<dim3(T_TOK / 16, NEXP), 256, 0, stream>>>(
        hs, nullptr, nullptr, nullptr, pWg, pWu, pWd,
        cnt, tokL, wL, nullptr, out);
  } else {
    // fallback: convert weights on the fly (workspace too small to repack)
    moe_ffn<true, false><<<dim3(T_TOK / 16), 256, 0, stream>>>(
        hs, sWg, sWu, sWd, nullptr, nullptr, nullptr,
        nullptr, nullptr, nullptr, sgate, out);
    moe_ffn<false, false><<<dim3(T_TOK / 16, NEXP), 256, 0, stream>>>(
        hs, Wg, Wu, Wd, nullptr, nullptr, nullptr,
        cnt, tokL, wL, nullptr, out);
  }
}
